// RepSLK_72224170049975
// MI455X (gfx1250) — compile-verified, matrix operand order checked by reading, not run on hardware
//
#include <hip/hip_runtime.h>
#include <hip/hip_bf16.h>
#include <math.h>

#define BB 16
#define CC 256
#define HH 64
#define WW 64
#define D4V 1024
#define PIX (HH*WW)      // 4096
#define NPIX (BB*PIX)    // 65536

typedef __attribute__((ext_vector_type(16))) __bf16 v16bf;
typedef __attribute__((ext_vector_type(8)))  float  v8f;
typedef __attribute__((ext_vector_type(4)))  unsigned int v4u;
typedef __attribute__((ext_vector_type(8)))  int v8i;
typedef __attribute__((ext_vector_type(4)))  int v4i;

#ifndef __has_builtin
#define __has_builtin(x) 0
#endif
#if __has_builtin(__builtin_amdgcn_tensor_load_to_lds) && \
    __has_builtin(__builtin_amdgcn_s_wait_tensorcnt)
#define USE_TDM 1
#else
#define USE_TDM 0
#endif

union FragAB { v16bf v; unsigned short u[16]; uint4 q[2]; };

__device__ __forceinline__ unsigned short f2bf(float f) {
    unsigned int u = __float_as_uint(f);
    u += 0x7FFFu + ((u >> 16) & 1u);      // round-to-nearest-even
    return (unsigned short)(u >> 16);
}
__device__ __forceinline__ float bf2f(unsigned short h) {
    return __uint_as_float(((unsigned int)h) << 16);
}
__device__ __forceinline__ FragAB load_frag(const unsigned short* p) {
    FragAB f;
    f.q[0] = *(const uint4*)(p);
    f.q[1] = *(const uint4*)(p + 16);
    return f;
}

#if USE_TDM
// Issue a TDM load described by D# groups 0/1 (<=2D tensor, groups 2/3 NULLed
// with zeros) and wait for completion. Must be executed by a single wave.
__device__ __forceinline__ void tdm_load_wait(v4u g0, v8i g1) {
    v4i gz = { 0, 0, 0, 0 };
#if defined(__clang_major__) && __clang_major__ >= 23
    v8i gz8 = { 0, 0, 0, 0, 0, 0, 0, 0 };
    __builtin_amdgcn_tensor_load_to_lds(g0, g1, gz, gz, gz8, 0);
#else
    __builtin_amdgcn_tensor_load_to_lds(g0, g1, gz, gz, 0);
#endif
    __builtin_amdgcn_s_wait_tensorcnt(0);
}
#endif

// ---------------------------------------------------------------------------
// Kernel 1: fold BN into the three depthwise kernels and merge into a cross
// stencil: Wv (31x5, center absorbs small 5x5 + lora2 center) and Wh (5x31
// with center 5x5 zeroed). Also zero the Gx^2 accumulator.
// ---------------------------------------------------------------------------
__global__ __launch_bounds__(256) void prep_kernel(
    const float* __restrict__ l1w, const float* __restrict__ l1g, const float* __restrict__ l1b,
    const float* __restrict__ l1m, const float* __restrict__ l1v,
    const float* __restrict__ l2w, const float* __restrict__ l2g, const float* __restrict__ l2b,
    const float* __restrict__ l2m, const float* __restrict__ l2v,
    const float* __restrict__ smw, const float* __restrict__ smg, const float* __restrict__ smb,
    const float* __restrict__ smm, const float* __restrict__ smv,
    float* __restrict__ Wv, float* __restrict__ Wh, float* __restrict__ biasC,
    float* __restrict__ Gx2)
{
    int c = threadIdx.x;
    float s1 = l1g[c] * rsqrtf(l1v[c] + 1e-5f);
    float s2 = l2g[c] * rsqrtf(l2v[c] + 1e-5f);
    float s3 = smg[c] * rsqrtf(smv[c] + 1e-5f);
    biasC[c] = (l1b[c] - l1m[c]*s1) + (l2b[c] - l2m[c]*s2) + (smb[c] - smm[c]*s3);

    for (int ky = 0; ky < 31; ++ky)
        for (int kx = 0; kx < 5; ++kx) {
            float wv = l1w[c*155 + ky*5 + kx] * s1;
            if (ky >= 13 && ky <= 17) {
                wv += smw[c*25 + (ky-13)*5 + kx] * s3;
                wv += l2w[c*155 + (ky-13)*31 + (kx+13)] * s2;
            }
            Wv[c*155 + ky*5 + kx] = wv;
        }
    for (int ky = 0; ky < 5; ++ky)
        for (int kx = 0; kx < 31; ++kx) {
            float wh = (kx >= 13 && kx <= 17) ? 0.f
                                              : l2w[c*155 + ky*31 + kx] * s2;
            Wh[c*155 + ky*31 + kx] = wh;
        }
    for (int i = c; i < BB*D4V; i += 256) Gx2[i] = 0.f;
}

// ---------------------------------------------------------------------------
// Kernel 2: convert pointwise weights to bf16 (B-matrix operands for WMMA)
// ---------------------------------------------------------------------------
__global__ __launch_bounds__(256) void cvt_kernel(
    const float* __restrict__ w1, const float* __restrict__ w2,
    unsigned short* __restrict__ w1bf, unsigned short* __restrict__ w2bf)
{
    int i = blockIdx.x * 256 + threadIdx.x;
    if (i < D4V*CC) { w1bf[i] = f2bf(w1[i]); w2bf[i] = f2bf(w2[i]); }
}

// ---------------------------------------------------------------------------
// Kernel 3: merged depthwise cross-stencil (285 taps), LDS-tiled, 4 outputs
// per thread with source-row register reuse. One block = (b, c, 16x64 strip).
// Output NCHW f32 scratch (perfectly coalesced stores).
// ---------------------------------------------------------------------------
__global__ __launch_bounds__(256) void dwconv_kernel(
    const float* __restrict__ x, const float* __restrict__ Wv,
    const float* __restrict__ Wh, const float* __restrict__ biasC,
    float* __restrict__ sOut)
{
    __shared__ __align__(16) float xt[46*94];
    __shared__ float wv[155];
    __shared__ float wh[155];
    int tid = threadIdx.x;
    int c = blockIdx.y, b = blockIdx.z, h0 = blockIdx.x * 16;

    if (tid < 155) { wv[tid] = Wv[c*155 + tid]; wh[tid] = Wh[c*155 + tid]; }

    const float* xc = x + ((size_t)(b*CC + c) << 12);
    for (int idx = tid; idx < 46*94; idx += 256) {
        int i = idx / 94, j = idx % 94;
        int gh = h0 - 15 + i, gw = j - 15;
        float v = 0.f;
        if (gh >= 0 && gh < HH && gw >= 0 && gw < WW) v = xc[gh*WW + gw];
        xt[idx] = v;
    }
    __syncthreads();

    int q = tid >> 6, ow = tid & 63;         // thread owns rows q*4..q*4+3, col ow
    float bias = biasC[c];
    float acc[4] = { bias, bias, bias, bias };

    // vertical strip 31x5: source rows q*4+sr, tap ky = sr - oy
    #pragma unroll
    for (int sr = 0; sr < 34; ++sr) {
        #pragma unroll
        for (int kx = 0; kx < 5; ++kx) {
            float v = xt[(q*4 + sr)*94 + ow + 13 + kx];
            #pragma unroll
            for (int oy = 0; oy < 4; ++oy) {
                int ky = sr - oy;
                if (ky >= 0 && ky < 31)
                    acc[oy] = fmaf(wv[ky*5 + kx], v, acc[oy]);
            }
        }
    }
    // horizontal strip 5x31 (center zeroed): source rows q*4+13+sr, ky = sr - oy
    #pragma unroll
    for (int sr = 0; sr < 8; ++sr) {
        #pragma unroll
        for (int kx = 0; kx < 31; ++kx) {
            float v = xt[(q*4 + 13 + sr)*94 + ow + kx];
            #pragma unroll
            for (int oy = 0; oy < 4; ++oy) {
                int ky = sr - oy;
                if (ky >= 0 && ky < 5)
                    acc[oy] = fmaf(wh[ky*31 + kx], v, acc[oy]);
            }
        }
    }

    float* so = sOut + ((size_t)(b*CC + c) << 12) + (h0 + q*4)*WW + ow;
    #pragma unroll
    for (int oy = 0; oy < 4; ++oy) so[oy*WW] = acc[oy];
}

// ---------------------------------------------------------------------------
// Kernel 4: fused LayerNorm + pwconv1 (bf16 WMMA) + exact GELU + Gx^2 partials.
// Block = 32 pixels x 1024 outputs; 8 waves x (2 M-tiles x 8 N-tiles).
// A-tile staged by TDM: 256 rows x 128B, row stride 16KB, with LDS padding
// (1 DWORD per 32) -> 33-float pitch makes LN column reads bank-conflict-free.
// ---------------------------------------------------------------------------
__global__ __launch_bounds__(256) void gemm1_kernel(
    const float* __restrict__ sIn, const unsigned short* __restrict__ w1bf,
    const float* __restrict__ b1, const float* __restrict__ ln_w,
    const float* __restrict__ ln_b, unsigned short* __restrict__ hOut,
    float* __restrict__ Gx2)
{
    __shared__ __align__(16) char smem[65536];
    float* sl2           = (float*)smem;                      // 33 KB: 256 x 33 f32 (padded)
    unsigned short* ybf  = (unsigned short*)(smem + 33792);   // 16 KB: 32 x 256 bf16
    // after a barrier, smem is reused as 8 x (32x128) bf16 h staging (64 KB)

    int tid = threadIdx.x, lane = tid & 31, wave = tid >> 5;
    int pixbase = blockIdx.x * 32;
    int b = pixbase >> 12;
    int pin = pixbase & 4095;

    // ---- stage A-tile [channel][pixel] into LDS ----
    unsigned long long gaddr =
        (unsigned long long)(const void*)(sIn + ((size_t)b*CC << 12) + pin);
#if USE_TDM
    if (wave == 0) {
        // D# group0: count=1, lds_addr=0 (single LDS object), global tile addr, type=2
        v4u g0 = { 1u, 0u, (unsigned int)gaddr,
                   (unsigned int)(gaddr >> 32) | 0x80000000u };
        // D# group1: data_size=8B, pad_enable, pad_interval=32dw, pad=1dw,
        // tensor_dim0=2048 (8B units), tensor_dim1=256, tile=16x256, stride=2048
        v8i g1 = { 0x01130000, 0x08000000, 0x01000000, 0x00100000,
                   256, 0x800, 0, 0 };
        tdm_load_wait(g0, g1);
    }
    __syncthreads();
#else
    {
        const float* sp = (const float*)(unsigned long long)gaddr + (size_t)tid*PIX;
        #pragma unroll
        for (int j = 0; j < 8; ++j) {
            float4 v = ((const float4*)sp)[j];
            sl2[tid*33 + j*4 + 0] = v.x;
            sl2[tid*33 + j*4 + 1] = v.y;
            sl2[tid*33 + j*4 + 2] = v.z;
            sl2[tid*33 + j*4 + 3] = v.w;
        }
    }
    __syncthreads();
#endif

    // LayerNorm over C=256 per pixel; each wave handles 4 pixels (wave32 reduce)
    #pragma unroll
    for (int pi = 0; pi < 4; ++pi) {
        int p = wave*4 + pi;
        float v[8]; float sum = 0.f, sq = 0.f;
        #pragma unroll
        for (int i = 0; i < 8; ++i) {
            v[i] = sl2[(lane + 32*i)*33 + p];
            sum += v[i]; sq += v[i]*v[i];
        }
        #pragma unroll
        for (int off = 16; off >= 1; off >>= 1) {
            sum += __shfl_xor(sum, off, 32);
            sq  += __shfl_xor(sq,  off, 32);
        }
        float mu   = sum * (1.f/256.f);
        float var  = sq  * (1.f/256.f) - mu*mu;
        float rstd = rsqrtf(var + 1e-6f);
        #pragma unroll
        for (int i = 0; i < 8; ++i) {
            int ch = lane + 32*i;
            ybf[p*256 + ch] = f2bf((v[i] - mu)*rstd*ln_w[ch] + ln_b[ch]);
        }
    }
    __syncthreads();

    int m = lane & 15, half = lane >> 4;
    int nwave = wave * 128;
    v8f zero = {0.f,0.f,0.f,0.f,0.f,0.f,0.f,0.f};
    v8f acc[16];
    #pragma unroll
    for (int t = 0; t < 16; ++t) acc[t] = zero;

    for (int kb = 0; kb < 8; ++kb) {
        int kbase = kb * 32;
        FragAB A0 = load_frag(&ybf[m*256 + kbase + half*8]);
        FragAB A1 = load_frag(&ybf[(16 + m)*256 + kbase + half*8]);
        const unsigned short* bcol =
            w1bf + (size_t)(nwave + m)*256 + kbase + half*8;
        FragAB Bcur = load_frag(bcol);
        #pragma unroll
        for (int t = 0; t < 8; ++t) {
            FragAB Bnext;
            if (t < 7) Bnext = load_frag(bcol + (size_t)(t + 1)*16*256);
            acc[t]     = __builtin_amdgcn_wmma_f32_16x16x32_bf16(
                false, A0.v, false, Bcur.v, (short)0, acc[t],     false, false);
            acc[8 + t] = __builtin_amdgcn_wmma_f32_16x16x32_bf16(
                false, A1.v, false, Bcur.v, (short)0, acc[8 + t], false, false);
            if (t < 7) Bcur = Bnext;
        }
    }
    __syncthreads();   // everyone done reading ybf -> smem reused as h staging

    // epilogue: +b1, exact GELU, bf16 stage, Gx^2 atomic partials
    unsigned short* hs = (unsigned short*)smem + wave*(32*128);
    #pragma unroll
    for (int t = 0; t < 8; ++t) {
        int d = nwave + t*16 + m;
        float bias = b1[d];
        float sq = 0.f;
        #pragma unroll
        for (int mt = 0; mt < 2; ++mt) {
            #pragma unroll
            for (int r = 0; r < 8; ++r) {
                float x0 = acc[mt*8 + t][r] + bias;
                float g  = 0.5f * x0 * (1.f + erff(x0 * 0.70710678118654752f));
                sq += g * g;
                hs[(mt*16 + r + 8*half)*128 + t*16 + m] = f2bf(g);
            }
        }
        sq += __shfl_xor(sq, 16, 32);
        if (half == 0) atomicAdd(&Gx2[b*D4V + d], sq);
    }

    // coalesced bf16 writeback of the wave's 32x128 tile
    unsigned short* hg = hOut + (size_t)pixbase * D4V + nwave;
    for (int e = lane; e < 512; e += 32) {
        int mr = e >> 4, j8 = (e & 15) * 8;
        *(uint4*)(hg + (size_t)mr*D4V + j8) = *(const uint4*)(hs + mr*128 + j8);
    }
}

// ---------------------------------------------------------------------------
// Kernel 5: finalize GRN: nscale[b,d] = grn_g[d]*Nx + 1
// ---------------------------------------------------------------------------
__global__ __launch_bounds__(256) void grn_kernel(
    const float* __restrict__ Gx2, const float* __restrict__ grn_g,
    float* __restrict__ nscale)
{
    __shared__ float red[256];
    int b = blockIdx.x, tid = threadIdx.x;
    float g[4]; float sum = 0.f;
    #pragma unroll
    for (int i = 0; i < 4; ++i) {
        g[i] = sqrtf(Gx2[b*D4V + tid + 256*i]);
        sum += g[i];
    }
    red[tid] = sum;
    __syncthreads();
    for (int off = 128; off >= 1; off >>= 1) {
        if (tid < off) red[tid] += red[tid + off];
        __syncthreads();
    }
    float inv = 1.f / (red[0] * (1.f/(float)D4V) + 1e-6f);
    #pragma unroll
    for (int i = 0; i < 4; ++i) {
        int d = tid + 256*i;
        nscale[b*D4V + d] = grn_g[d] * (g[i] * inv) + 1.f;
    }
}

// ---------------------------------------------------------------------------
// Kernel 6: fused GRN-apply + pwconv2 (bf16 WMMA) + b2 + residual, NCHW out.
// Block = 32 pixels x 256 outputs; 8 waves x (2 M-tiles x 2 N-tiles).
// A-tile (64KB contiguous bf16) staged by a 1D TDM load, then GRN scale
// applied as an in-place LDS RMW.
// ---------------------------------------------------------------------------
__global__ __launch_bounds__(256) void gemm2_kernel(
    const unsigned short* __restrict__ hIn, const unsigned short* __restrict__ w2bf,
    const float* __restrict__ nscale, const float* __restrict__ grn_b,
    const float* __restrict__ b2, const float* __restrict__ x,
    float* __restrict__ out)
{
    __shared__ __align__(16) unsigned short hbf[32*1024];   // 64 KB (single LDS object)
    int tid = threadIdx.x, lane = tid & 31, wave = tid >> 5;
    int pixbase = blockIdx.x * 32;
    int b = pixbase >> 12;
    int pin = pixbase & 4095;
    int m = lane & 15, half = lane >> 4;
    int cwave = wave * 32;

    // warm L2/WGP$ for the residual rows this lane will need in the epilogue
    {
        size_t base0 = (((size_t)(b*CC + cwave + m)) << 12) + pin;
        __builtin_prefetch((const void*)(x + base0), 0, 1);
        __builtin_prefetch((const void*)(x + base0 + ((size_t)16 << 12)), 0, 1);
    }

    const unsigned short* hp = hIn + (size_t)pixbase * D4V;
#if USE_TDM
    {
        unsigned long long gaddr = (unsigned long long)(const void*)hp;
        if (wave == 0) {
            // 1D contiguous 64KB: data_size=8B, tensor_dim0=tile_dim0=8192
            v4u g0 = { 1u, 0u, (unsigned int)gaddr,
                       (unsigned int)(gaddr >> 32) | 0x80000000u };
            v8i g1 = { 0x00030000, 0x20000000, 0, 0x20000000, 0, 0, 0, 0 };
            tdm_load_wait(g0, g1);
        }
        __syncthreads();
        // in-place GRN scale: h' = h*(g*Nx+1) + grn_b  (disjoint uint4 per thread)
        const float* ns = nscale + b*D4V;
        for (int i8 = tid; i8 < 4096; i8 += 256) {
            uint4 hv = ((const uint4*)hbf)[i8];
            int dbase = (i8 & 127) * 8;
            unsigned int wi[4] = { hv.x, hv.y, hv.z, hv.w };
            unsigned int wo[4];
            #pragma unroll
            for (int k = 0; k < 4; ++k) {
                int d0 = dbase + k*2;
                float lo = bf2f((unsigned short)(wi[k] & 0xFFFFu));
                float hi = bf2f((unsigned short)(wi[k] >> 16));
                lo = lo * ns[d0]     + grn_b[d0];
                hi = hi * ns[d0 + 1] + grn_b[d0 + 1];
                wo[k] = (unsigned int)f2bf(lo) | ((unsigned int)f2bf(hi) << 16);
            }
            uint4 ov = { wo[0], wo[1], wo[2], wo[3] };
            ((uint4*)hbf)[i8] = ov;
        }
    }
#else
    {
        const float* ns = nscale + b*D4V;
        for (int i8 = tid; i8 < 4096; i8 += 256) {
            uint4 hv = ((const uint4*)hp)[i8];
            int dbase = (i8 & 127) * 8;
            unsigned int wi[4] = { hv.x, hv.y, hv.z, hv.w };
            unsigned int wo[4];
            #pragma unroll
            for (int k = 0; k < 4; ++k) {
                int d0 = dbase + k*2;
                float lo = bf2f((unsigned short)(wi[k] & 0xFFFFu));
                float hi = bf2f((unsigned short)(wi[k] >> 16));
                lo = lo * ns[d0]     + grn_b[d0];
                hi = hi * ns[d0 + 1] + grn_b[d0 + 1];
                wo[k] = (unsigned int)f2bf(lo) | ((unsigned int)f2bf(hi) << 16);
            }
            uint4 ov = { wo[0], wo[1], wo[2], wo[3] };
            ((uint4*)hbf)[i8] = ov;
        }
    }
#endif
    __syncthreads();

    v8f zero = {0.f,0.f,0.f,0.f,0.f,0.f,0.f,0.f};
    v8f acc[4] = { zero, zero, zero, zero };   // [t*2 + mt]

    const unsigned short* bbase = w2bf + (size_t)(cwave + m)*D4V + half*8;
    for (int kb = 0; kb < 32; ++kb) {
        int kbase = kb * 32;
        FragAB A0 = load_frag(&hbf[m*1024 + kbase + half*8]);
        FragAB A1 = load_frag(&hbf[(16 + m)*1024 + kbase + half*8]);
        FragAB B0 = load_frag(bbase + kbase);
        FragAB B1 = load_frag(bbase + (size_t)16*D4V + kbase);
        acc[0] = __builtin_amdgcn_wmma_f32_16x16x32_bf16(
            false, A0.v, false, B0.v, (short)0, acc[0], false, false);
        acc[1] = __builtin_amdgcn_wmma_f32_16x16x32_bf16(
            false, A1.v, false, B0.v, (short)0, acc[1], false, false);
        acc[2] = __builtin_amdgcn_wmma_f32_16x16x32_bf16(
            false, A0.v, false, B1.v, (short)0, acc[2], false, false);
        acc[3] = __builtin_amdgcn_wmma_f32_16x16x32_bf16(
            false, A1.v, false, B1.v, (short)0, acc[3], false, false);
    }

    #pragma unroll
    for (int t = 0; t < 2; ++t) {
        int c = cwave + t*16 + m;
        float bias = b2[c];
        #pragma unroll
        for (int mt = 0; mt < 2; ++mt) {
            v8f a = acc[t*2 + mt];
            size_t base = (((size_t)(b*CC + c)) << 12) + pin + mt*16 + half*8;
            const float4* xr = (const float4*)(x + base);
            float4 x0 = xr[0], x1 = xr[1];
            float4 r0, r1;
            r0.x = a[0] + bias + x0.x;  r0.y = a[1] + bias + x0.y;
            r0.z = a[2] + bias + x0.z;  r0.w = a[3] + bias + x0.w;
            r1.x = a[4] + bias + x1.x;  r1.y = a[5] + bias + x1.y;
            r1.z = a[6] + bias + x1.z;  r1.w = a[7] + bias + x1.w;
            ((float4*)(out + base))[0] = r0;
            ((float4*)(out + base))[1] = r1;
        }
    }
}

// ---------------------------------------------------------------------------
extern "C" void kernel_launch(void* const* d_in, const int* in_sizes, int n_in,
                              void* d_out, int out_size, void* d_ws, size_t ws_size,
                              hipStream_t stream)
{
    (void)in_sizes; (void)n_in; (void)out_size; (void)ws_size;
    const float* x     = (const float*)d_in[0];
    const float* l1w   = (const float*)d_in[1];
    const float* l1g   = (const float*)d_in[2];
    const float* l1b   = (const float*)d_in[3];
    const float* l1m   = (const float*)d_in[4];
    const float* l1v   = (const float*)d_in[5];
    const float* l2w   = (const float*)d_in[6];
    const float* l2g   = (const float*)d_in[7];
    const float* l2b   = (const float*)d_in[8];
    const float* l2m   = (const float*)d_in[9];
    const float* l2v   = (const float*)d_in[10];
    const float* smw   = (const float*)d_in[11];
    const float* smg   = (const float*)d_in[12];
    const float* smb   = (const float*)d_in[13];
    const float* smm   = (const float*)d_in[14];
    const float* smv   = (const float*)d_in[15];
    const float* ln_w  = (const float*)d_in[16];
    const float* ln_b  = (const float*)d_in[17];
    const float* w1    = (const float*)d_in[18];
    const float* b1    = (const float*)d_in[19];
    const float* grn_g = (const float*)d_in[20];
    const float* grn_b = (const float*)d_in[21];
    const float* w2    = (const float*)d_in[22];
    const float* b2    = (const float*)d_in[23];

    char* ws = (char*)d_ws;
    float*          Wv     = (float*)(ws + 0);                 // 158720 B
    float*          Wh     = (float*)(ws + 158720);            // 158720 B
    float*          biasC  = (float*)(ws + 317440);            // 1024 B
    unsigned short* w1bf   = (unsigned short*)(ws + 318464);   // 512 KB
    unsigned short* w2bf   = (unsigned short*)(ws + 842752);   // 512 KB
    float*          Gx2    = (float*)(ws + 1367040);           // 64 KB
    float*          nscale = (float*)(ws + 1432576);           // 64 KB
    float*          sbuf   = (float*)(ws + 1498112);           // 64 MB (NCHW f32)
    unsigned short* hbuf   = (unsigned short*)(ws + 68606976); // 128 MB (bf16)

    prep_kernel<<<1, 256, 0, stream>>>(l1w, l1g, l1b, l1m, l1v,
                                       l2w, l2g, l2b, l2m, l2v,
                                       smw, smg, smb, smm, smv,
                                       Wv, Wh, biasC, Gx2);
    cvt_kernel<<<1024, 256, 0, stream>>>(w1, w2, w1bf, w2bf);
    dwconv_kernel<<<dim3(4, 256, 16), 256, 0, stream>>>(x, Wv, Wh, biasC, sbuf);
    gemm1_kernel<<<NPIX/32, 256, 0, stream>>>(sbuf, w1bf, b1, ln_w, ln_b, hbuf, Gx2);
    grn_kernel<<<16, 256, 0, stream>>>(Gx2, grn_g, nscale);
    gemm2_kernel<<<NPIX/32, 256, 0, stream>>>(hbuf, w2bf, nscale, grn_b, b2, x,
                                              (float*)d_out);
}
